// OptimizedMPNN_39273180955640
// MI455X (gfx1250) — 8-bit matrix operand order repaired and checked by reading, compiled, not run on hardware
//
#include <hip/hip_runtime.h>
#include <cstdint>

#define NN 20000
#define EE 50000
#define INC 32
#define ECC 8
#define HH 64
#define GG 128

typedef __attribute__((ext_vector_type(16))) int          v16i;
typedef __attribute__((ext_vector_type(8)))  float        v8f;
typedef __attribute__((ext_vector_type(4)))  unsigned int v4u;
typedef __attribute__((ext_vector_type(8)))  int          v8i_t;
typedef __attribute__((ext_vector_type(4)))  int          v4i_t;

__device__ inline unsigned char f32_to_fp8_e4m3(float f) {
    unsigned char s = (f < 0.f) ? 0x80 : 0x00;
    float a = fabsf(f);
    if (!(a > 0.f)) return s;
    if (a >= 448.f) return s | 0x7E;
    int e;
    float m = frexpf(a, &e);
    int E = e + 6;
    if (E >= 1) {
        int q = (int)(m * 16.f + 0.5f);
        if (q == 16) { q = 8; ++E; }
        if (E > 15) return s | 0x7E;
        return s | (unsigned char)((E << 3) | (q - 8));
    } else {
        int q = (int)(a * 512.f + 0.5f);
        if (q > 7) q = 7;
        return s | (unsigned char)q;
    }
}

__global__ void fill_kernel(float* p, int n, float v) {
    int i = blockIdx.x * blockDim.x + threadIdx.x;
    if (i < n) p[i] = v;
}

__global__ void repack_b_fp8_kernel(const float* __restrict__ w2,
                                    unsigned char* __restrict__ Bp,
                                    int chunks, int ncol) {
    int idx = blockIdx.x * blockDim.x + threadIdx.x;
    int total = chunks * ncol * 128;
    if (idx >= total) return;
    int c    = idx / (ncol * 128);
    int rem  = idx % (ncol * 128);
    int col  = rem / 128;
    int jj   = rem % 128;
    int half = jj >> 6;
    int b    = jj & 63;
    int q    = b >> 4;
    int r    = b & 15;
    int K    = c * 128 + q * 32 + r + half * 16;
    Bp[idx] = f32_to_fp8_e4m3(w2[(size_t)K * ncol + col]);
}

__global__ void edge_hidden_fp8_kernel(const float* __restrict__ ea, const float* __restrict__ w1,
                                       const float* __restrict__ b1, unsigned char* __restrict__ h,
                                       int hid) {
    int idx = blockIdx.x * blockDim.x + threadIdx.x;
    if (idx >= EE * hid) return;
    int e = idx / hid;
    int k = idx % hid;
    float acc = b1[k];
    #pragma unroll
    for (int c = 0; c < ECC; ++c) acc += ea[e * ECC + c] * w1[c * hid + k];
    h[idx] = f32_to_fp8_e4m3(fmaxf(acc, 0.f));
}

__global__ void edge_count_kernel(const int* __restrict__ dst, float* __restrict__ cnt) {
    int e = blockIdx.x * blockDim.x + threadIdx.x;
    if (e < EE) atomicAdd(&cnt[dst[e]], 1.0f);
}

template <int CHUNKS, int CIN, int NCOL>
__global__ void __launch_bounds__(128)
nnconv_msg_fp8_kernel(const unsigned char* __restrict__ hbuf,
                      const unsigned char* __restrict__ Bp,
                      const float* __restrict__ b2,
                      const float* __restrict__ feats,
                      const int* __restrict__ src, const int* __restrict__ dst,
                      float* __restrict__ sOut) {
    constexpr int KHT = CHUNKS * 128;
    __shared__ unsigned char ldsH[32 * KHT];
    __shared__ float ldsXT[CIN * 32];
    __shared__ float ldsB2[NCOL];
    __shared__ int   ldsSrc[32];
    __shared__ int   ldsDst[32];

    const int tid = threadIdx.x;
    const int te  = blockIdx.x * 32;

    if (tid < 32) {
        const unsigned long long ga = (unsigned long long)(hbuf + (size_t)te * KHT);
        const unsigned int rows = (unsigned int)(EE - te);
        v4u g0;
        g0[0] = 1u;
        g0[1] = (unsigned int)(uintptr_t)&ldsH[0];
        g0[2] = (unsigned int)ga;
        g0[3] = (unsigned int)((ga >> 32) & 0x01FFFFFFu)
              | 0x80000000u;
        v8i_t g1;
        g1[0] = 0;
        g1[1] = (int)(((unsigned)KHT & 0xFFFFu) << 16);
        g1[2] = (int)((rows & 0xFFFFu) << 16);
        g1[3] = (int)(((unsigned)KHT & 0xFFFFu) << 16);
        g1[4] = 32;
        g1[5] = KHT;
        g1[6] = 0;
        g1[7] = 0;
        v4i_t g2 = {0, 0, 0, 0};
        v4i_t g3 = {0, 0, 0, 0};
#if defined(__clang_major__) && (__clang_major__ >= 23)
        v8i_t g4 = {0, 0, 0, 0, 0, 0, 0, 0};
        __builtin_amdgcn_tensor_load_to_lds(g0, g1, g2, g3, g4, 0);
#else
        __builtin_amdgcn_tensor_load_to_lds(g0, g1, g2, g3, 0);
#endif
        __builtin_amdgcn_s_wait_tensorcnt(0);
    } else {
        if (tid < 64) {
            int t = tid - 32;
            int ge = te + t;
            ldsSrc[t] = (ge < EE) ? src[ge] : 0;
            ldsDst[t] = (ge < EE) ? dst[ge] : 0;
        }
        for (int t = tid - 32; t < NCOL; t += 96) ldsB2[t] = b2[t];
    }
    __syncthreads();
    for (int t = tid; t < 32 * CIN; t += 128) {
        int e = t / CIN, i = t % CIN;
        ldsXT[i * 32 + e] = (te + e < EE) ? feats[(size_t)ldsSrc[e] * CIN + i] : 0.f;
    }
    __syncthreads();

    const int wv   = tid >> 5;
    const int lane = tid & 31;
    const int m    = lane & 15;
    const int hi8  = (lane >= 16) ? 8 : 0;
    const int bko  = (lane >= 16) ? 64 : 0;

    v16i af0[CHUNKS], af1[CHUNKS];
    #pragma unroll
    for (int c = 0; c < CHUNKS; ++c) {
        #pragma unroll
        for (int d = 0; d < 16; ++d) {
            unsigned int w0 = 0, w1 = 0;
            #pragma unroll
            for (int bb = 0; bb < 4; ++bb) {
                int jj  = d * 4 + bb;
                int blk = jj >> 5;
                int j   = jj & 31;
                int K   = c * 128 + blk * 64 + j + ((j >> 4) << 4) + (hi8 << 1);
                w0 |= (unsigned int)ldsH[m * KHT + K] << (8 * bb);
                w1 |= (unsigned int)ldsH[(16 + m) * KHT + K] << (8 * bb);
            }
            af0[c][d] = (int)w0;
            af1[c][d] = (int)w1;
        }
    }

    v8f msg0 = {}, msg1 = {};
    for (int i = 0; i < CIN; ++i) {
        const int col = i * 64 + wv * 16 + m;
        if (i + 1 < CIN)
            __builtin_prefetch(Bp + ((size_t)(col + 64)) * 128 + bko, 0, 3);
        v8f acc0 = {}, acc1 = {};
        #pragma unroll
        for (int c = 0; c < CHUNKS; ++c) {
            const v16i bf = *reinterpret_cast<const v16i*>(
                Bp + ((size_t)c * NCOL + col) * 128 + bko);
            acc0 = __builtin_amdgcn_wmma_f32_16x16x128_fp8_fp8(af0[c], bf, (short)0, acc0,
                                                               false, false);
            acc1 = __builtin_amdgcn_wmma_f32_16x16x128_fp8_fp8(af1[c], bf, (short)0, acc1,
                                                               false, false);
        }
        const float bc = ldsB2[col];
        #pragma unroll
        for (int r = 0; r < 8; ++r) {
            msg0[r] += ldsXT[i * 32 + (r + hi8)] * (acc0[r] + bc);
            msg1[r] += ldsXT[i * 32 + (16 + r + hi8)] * (acc1[r] + bc);
        }
    }

    const int o = wv * 16 + m;
    #pragma unroll
    for (int r = 0; r < 8; ++r) {
        int row0 = r + hi8, row1 = 16 + r + hi8;
        if (te + row0 < EE) atomicAdd(&sOut[(size_t)ldsDst[row0] * 64 + o], msg0[r]);
        if (te + row1 < EE) atomicAdd(&sOut[(size_t)ldsDst[row1] * 64 + o], msg1[r]);
    }
}

__global__ void node_update_kernel(const float* __restrict__ s, const float* __restrict__ cnt,
                                   const float* __restrict__ feats, const float* __restrict__ root,
                                   const float* __restrict__ bias, float* __restrict__ z,
                                   float* __restrict__ bnsum, float* __restrict__ bnsq, int cin) {
    int idx = blockIdx.x * blockDim.x + threadIdx.x;
    if (idx >= NN * HH) return;
    int n = idx / HH, o = idx % HH;
    float acc = s[idx] / fmaxf(cnt[n], 1.0f) + bias[o];
    for (int i = 0; i < cin; ++i) acc += feats[n * cin + i] * root[i * HH + o];
    z[idx] = acc;
    atomicAdd(&bnsum[o], acc);
    atomicAdd(&bnsq[o], acc * acc);
}

__global__ void bn_relu_kernel(const float* __restrict__ z, const float* __restrict__ bnsum,
                               const float* __restrict__ bnsq, const float* __restrict__ g,
                               const float* __restrict__ b, float* __restrict__ out) {
    int idx = blockIdx.x * blockDim.x + threadIdx.x;
    if (idx >= NN * HH) return;
    int o = idx % HH;
    float mean = bnsum[o] / (float)NN;
    float var  = bnsq[o] / (float)NN - mean * mean;
    float v = g[o] * (z[idx] - mean) * rsqrtf(var + 1e-5f) + b[o];
    out[idx] = fmaxf(v, 0.f);
}

__global__ void bn_relu_pool_kernel(const float* __restrict__ z, const float* __restrict__ bnsum,
                                    const float* __restrict__ bnsq, const float* __restrict__ g,
                                    const float* __restrict__ b, const int* __restrict__ batch,
                                    float* __restrict__ gadd, float* __restrict__ gcnt) {
    int idx = blockIdx.x * blockDim.x + threadIdx.x;
    if (idx >= NN * HH) return;
    int n = idx / HH, o = idx % HH;
    float mean = bnsum[o] / (float)NN;
    float var  = bnsq[o] / (float)NN - mean * mean;
    float v = fmaxf(g[o] * (z[idx] - mean) * rsqrtf(var + 1e-5f) + b[o], 0.f);
    int gi = batch[n];
    atomicAdd(&gadd[(size_t)gi * HH + o], v);
    if (o == 0) atomicAdd(&gcnt[gi], 1.0f);
}

__global__ void head_kernel(const float* __restrict__ gadd, const float* __restrict__ gcnt,
                            const float* __restrict__ w1, const float* __restrict__ b1,
                            const float* __restrict__ w2, const float* __restrict__ b2,
                            const float* __restrict__ w3, const float* __restrict__ b3,
                            float* __restrict__ out) {
    __shared__ float v[2 * HH];
    __shared__ float a1[HH];
    __shared__ float a2[HH / 2];
    int g = blockIdx.x, t = threadIdx.x;
    float inv = 1.0f / fmaxf(gcnt[g], 1.0f);
    float add = gadd[(size_t)g * HH + t];
    v[t] = add * inv;
    v[HH + t] = add;
    __syncthreads();
    float acc = b1[t];
    for (int i = 0; i < 2 * HH; ++i) acc += v[i] * w1[i * HH + t];
    a1[t] = fmaxf(acc, 0.f);
    __syncthreads();
    if (t < HH / 2) {
        float a = b2[t];
        for (int j = 0; j < HH; ++j) a += a1[j] * w2[j * (HH / 2) + t];
        a2[t] = fmaxf(a, 0.f);
    }
    __syncthreads();
    if (t == 0) {
        float a = b3[0];
        for (int k = 0; k < HH / 2; ++k) a += a2[k] * w3[k];
        out[g] = a;
    }
}

static inline size_t alignup(size_t x) { return (x + 255) & ~(size_t)255; }

extern "C" void kernel_launch(void* const* d_in, const int* in_sizes, int n_in,
                              void* d_out, int out_size, void* d_ws, size_t ws_size,
                              hipStream_t stream) {
    (void)in_sizes; (void)n_in; (void)out_size; (void)ws_size;
    const float* x     = (const float*)d_in[0];
    const int*   eidx  = (const int*)d_in[1];
    const float* eattr = (const float*)d_in[2];
    const int*   batch = (const int*)d_in[3];
    const float* m1w1  = (const float*)d_in[4];
    const float* m1b1  = (const float*)d_in[5];
    const float* m1w2  = (const float*)d_in[6];
    const float* m1b2  = (const float*)d_in[7];
    const float* root1 = (const float*)d_in[8];
    const float* bias1 = (const float*)d_in[9];
    const float* bn1g  = (const float*)d_in[10];
    const float* bn1b  = (const float*)d_in[11];
    const float* m2w1  = (const float*)d_in[12];
    const float* m2b1  = (const float*)d_in[13];
    const float* m2w2  = (const float*)d_in[14];
    const float* m2b2  = (const float*)d_in[15];
    const float* root2 = (const float*)d_in[16];
    const float* bias2 = (const float*)d_in[17];
    const float* bn2g  = (const float*)d_in[18];
    const float* bn2b  = (const float*)d_in[19];
    const float* fc1w  = (const float*)d_in[20];
    const float* fc1b  = (const float*)d_in[21];
    const float* fc2w  = (const float*)d_in[22];
    const float* fc2b  = (const float*)d_in[23];
    const float* fc3w  = (const float*)d_in[24];
    const float* fc3b  = (const float*)d_in[25];
    const int* src = eidx;
    const int* dst = eidx + EE;
    float* out = (float*)d_out;

    char* ws = (char*)d_ws;
    size_t off = 0;
    unsigned char* B1 = (unsigned char*)(ws + off); off = alignup(off + (size_t)1 * 2048 * 128);
    unsigned char* B2 = (unsigned char*)(ws + off); off = alignup(off + (size_t)2 * 4096 * 128);
    unsigned char* H1 = (unsigned char*)(ws + off); off = alignup(off + (size_t)EE * 128);
    unsigned char* H2 = (unsigned char*)(ws + off); off = alignup(off + (size_t)EE * 256);
    float* S     = (float*)(ws + off); off = alignup(off + (size_t)NN * HH * 4);
    float* CNT   = (float*)(ws + off); off = alignup(off + (size_t)NN * 4);
    float* Z     = (float*)(ws + off); off = alignup(off + (size_t)NN * HH * 4);
    float* HNODE = (float*)(ws + off); off = alignup(off + (size_t)NN * HH * 4);
    float* BNS   = (float*)(ws + off); off = alignup(off + 128 * 4);
    float* GADD  = (float*)(ws + off); off = alignup(off + (size_t)GG * HH * 4);
    float* GCNT  = (float*)(ws + off); off = alignup(off + (size_t)GG * 4);

    auto fill = [&](float* p, int n) {
        fill_kernel<<<(n + 255) / 256, 256, 0, stream>>>(p, n, 0.f);
    };

    fill(S, NN * HH); fill(CNT, NN); fill(BNS, 128); fill(GADD, GG * HH); fill(GCNT, GG);

    repack_b_fp8_kernel<<<(1 * 2048 * 128 + 255) / 256, 256, 0, stream>>>(m1w2, B1, 1, 2048);
    repack_b_fp8_kernel<<<(2 * 4096 * 128 + 255) / 256, 256, 0, stream>>>(m2w2, B2, 2, 4096);

    edge_hidden_fp8_kernel<<<(EE * 128 + 255) / 256, 256, 0, stream>>>(eattr, m1w1, m1b1, H1, 128);
    edge_count_kernel<<<(EE + 255) / 256, 256, 0, stream>>>(dst, CNT);
    nnconv_msg_fp8_kernel<1, 32, 2048><<<(EE + 31) / 32, 128, 0, stream>>>(H1, B1, m1b2, x,
                                                                           src, dst, S);
    node_update_kernel<<<(NN * HH + 255) / 256, 256, 0, stream>>>(S, CNT, x, root1, bias1, Z,
                                                                  BNS, BNS + 64, INC);
    bn_relu_kernel<<<(NN * HH + 255) / 256, 256, 0, stream>>>(Z, BNS, BNS + 64, bn1g, bn1b, HNODE);

    fill(S, NN * HH); fill(BNS, 128);
    edge_hidden_fp8_kernel<<<(EE * 256 + 255) / 256, 256, 0, stream>>>(eattr, m2w1, m2b1, H2, 256);
    nnconv_msg_fp8_kernel<2, 64, 4096><<<(EE + 31) / 32, 128, 0, stream>>>(H2, B2, m2b2, HNODE,
                                                                           src, dst, S);
    node_update_kernel<<<(NN * HH + 255) / 256, 256, 0, stream>>>(S, CNT, HNODE, root2, bias2, Z,
                                                                  BNS, BNS + 64, HH);
    bn_relu_pool_kernel<<<(NN * HH + 255) / 256, 256, 0, stream>>>(Z, BNS, BNS + 64, bn2g, bn2b,
                                                                   batch, GADD, GCNT);

    head_kernel<<<GG, HH, 0, stream>>>(GADD, GCNT, fc1w, fc1b, fc2w, fc2b, fc3w, fc3b, out);
}
